// CausalSelfAttention_26603027431778
// MI455X (gfx1250) — matrix operand order repaired and checked by reading, not compiled, not run on hardware
//
#include <hip/hip_runtime.h>
#include <hip/hip_bf16.h>

// ---------------------------------------------------------------------------
// CausalSelfAttention for MI455X (gfx1250, wave32, WMMA).
// All matmuls via v_wmma_f32_16x16x32_f16 (f16 in, f32 accumulate).
// Softmax fully in fp32 with online (flash) rescaling.
// All dims compile-time: B=4, T=2048, D=1024, H=16, Dh=64.
// ---------------------------------------------------------------------------

#define D_MODEL 1024
#define N_HEADS 16
#define D_HEAD  64
#define SEQ_T   2048
#define BATCH   4
#define M_ROWS  (BATCH * SEQ_T)   // 8192
#define KT32    (D_MODEL / 32)    // 32 k-tiles of 32

typedef __attribute__((ext_vector_type(16))) _Float16 v16h;
typedef __attribute__((ext_vector_type(8)))  _Float16 v8h;
typedef __attribute__((ext_vector_type(8)))  float    v8f;

// 16-bit A-matrix 16x32 fragment, this code's contiguous k order (not ISA table order). lane m = lane&15.
// element e -> K = lanehi*16 + e   (matches B fragment order)
__device__ __host__ __forceinline__ int amap_k(int lane, int e) {
  return (((lane >> 4) & 1) << 4) | e;
}
// 16-bit B-matrix 32x16 fragment. lane n = lane&15. element e -> K = lanehi*16 + e
__device__ __host__ __forceinline__ int bmap_k(int lane, int e) {
  return (((lane >> 4) & 1) << 4) | e;
}

__device__ __forceinline__ v16h join8(v8h lo, v8h hi) {
  return __builtin_shufflevector(lo, hi, 0, 1, 2, 3, 4, 5, 6, 7,
                                 8, 9, 10, 11, 12, 13, 14, 15);
}

// ---------------------------------------------------------------------------
// Pack kernels: fragment-ready f16 buffers.
// Layout: tile (mt,kt): halves [ ((mt*KT32+kt)*32 + lane)*16 + e ]  -> each
// lane's 16-half fragment is one contiguous, 32B-aligned chunk.
// ---------------------------------------------------------------------------
__global__ void pack_a_f32(const float* __restrict__ src, _Float16* __restrict__ dst) {
  int idx  = blockIdx.x * blockDim.x + threadIdx.x;
  int e    = idx & 15;
  int lane = (idx >> 4) & 31;
  int tile = idx >> 9;
  int kt   = tile & (KT32 - 1);
  int mt   = tile >> 5;
  int m = (mt << 4) | (lane & 15);
  int k = (kt << 5) + amap_k(lane, e);
  dst[idx] = (_Float16)src[((size_t)m << 10) + k];
}

__global__ void pack_a_f16(const _Float16* __restrict__ src, _Float16* __restrict__ dst) {
  int idx  = blockIdx.x * blockDim.x + threadIdx.x;
  int e    = idx & 15;
  int lane = (idx >> 4) & 31;
  int tile = idx >> 9;
  int kt   = tile & (KT32 - 1);
  int mt   = tile >> 5;
  int m = (mt << 4) | (lane & 15);
  int k = (kt << 5) + amap_k(lane, e);
  dst[idx] = src[((size_t)m << 10) + k];
}

// Weights (N_out, K_in) row-major; y = x @ W^T -> B[k][n] = W[n][k] = rows of W.
__global__ void pack_b_f32(const float* __restrict__ src, _Float16* __restrict__ dst) {
  int idx  = blockIdx.x * blockDim.x + threadIdx.x;
  int e    = idx & 15;
  int lane = (idx >> 4) & 31;
  int tile = idx >> 9;
  int kt   = tile & (KT32 - 1);
  int nt   = tile >> 5;
  int n = (nt << 4) | (lane & 15);
  int k = (kt << 5) + bmap_k(lane, e);
  dst[idx] = (_Float16)src[((size_t)n << 10) + k];
}

// ---------------------------------------------------------------------------
// GEMM: C(8192x1024) = A @ B^T, K=1024, pre-packed f16 fragments.
// 256 threads = 8 waves; block tile 128x128; each wave 32x64 (2x4 WMMA tiles).
// Loads are (uniform SGPR offset) + (lane*32B) so address math stays scalar.
// mode 0: fp32 row-major C.  mode 1: f16 (B,H,T,Dh) layout for Q/K/V.
// ---------------------------------------------------------------------------
__global__ __launch_bounds__(256)
void gemm_wmma(const _Float16* __restrict__ A, const _Float16* __restrict__ Bw,
               float* __restrict__ Cf32, _Float16* __restrict__ Cqkv, int mode) {
  const int lane = threadIdx.x & 31;
  const int wave = threadIdx.x >> 5;
  const int wm = wave >> 1, wn = wave & 1;
  const int mt0 = blockIdx.y * 8 + wm * 2;   // 2 M-tiles per wave
  const int nt0 = blockIdx.x * 8 + wn * 4;   // 4 N-tiles per wave
  const int laneoff = lane * 16;             // halves; 32B per lane

  v8f c[2][4];
  const v8f vzero = {0.f, 0.f, 0.f, 0.f, 0.f, 0.f, 0.f, 0.f};
#pragma unroll
  for (int i = 0; i < 2; ++i)
#pragma unroll
    for (int j = 0; j < 4; ++j) c[i][j] = vzero;

  for (int kt = 0; kt < KT32; ++kt) {
    v16h a[2], b[4];
#pragma unroll
    for (int i = 0; i < 2; ++i) {
      size_t uoff = (size_t)(((mt0 + i) << 5) + kt) << 9;  // tile*512 halves
      a[i] = *(const v16h*)(A + uoff + laneoff);
    }
#pragma unroll
    for (int j = 0; j < 4; ++j) {
      size_t uoff = (size_t)(((nt0 + j) << 5) + kt) << 9;
      b[j] = *(const v16h*)(Bw + uoff + laneoff);
    }
#pragma unroll
    for (int i = 0; i < 2; ++i)
#pragma unroll
      for (int j = 0; j < 4; ++j)
        c[i][j] = __builtin_amdgcn_wmma_f32_16x16x32_f16(
            false, a[i], false, b[j], (short)0, c[i][j], false, false);
  }

  // C/D layout: VGPR r holds row r (lanes 0-15) / r+8 (lanes 16-31); col = lane&15.
#pragma unroll
  for (int i = 0; i < 2; ++i)
#pragma unroll
    for (int j = 0; j < 4; ++j)
#pragma unroll
      for (int r = 0; r < 8; ++r) {
        int m = ((mt0 + i) << 4) + r + ((lane >> 4) << 3);
        int n = ((nt0 + j) << 4) + (lane & 15);
        float v = c[i][j][r];
        if (mode == 0) {
          Cf32[((size_t)m << 10) + n] = v;
        } else {
          int bb = m >> 11, t = m & (SEQ_T - 1);
          int h = n >> 6, d = n & (D_HEAD - 1);
          Cqkv[(((size_t)(bb * N_HEADS + h) << 11) + t) * D_HEAD + d] = (_Float16)v;
        }
      }
}

// ---------------------------------------------------------------------------
// Flash attention: one workgroup = 4 waves = 64 query rows of one (b,h).
// Waves stream causal key chunks of 32 independently (no block barriers).
// ---------------------------------------------------------------------------
__device__ __forceinline__ float xor_red_max16(float v) {
  v = fmaxf(v, __shfl_xor(v, 1));
  v = fmaxf(v, __shfl_xor(v, 2));
  v = fmaxf(v, __shfl_xor(v, 4));
  v = fmaxf(v, __shfl_xor(v, 8));
  return v;
}
__device__ __forceinline__ float xor_red_sum16(float v) {
  v += __shfl_xor(v, 1);
  v += __shfl_xor(v, 2);
  v += __shfl_xor(v, 4);
  v += __shfl_xor(v, 8);
  return v;
}

__global__ __launch_bounds__(128)
void attn_flash(const _Float16* __restrict__ Q, const _Float16* __restrict__ K,
                const _Float16* __restrict__ V, _Float16* __restrict__ Oc) {
  __shared__ _Float16 pbuf[4 * 16 * 32];  // per-wave P transpose staging (1KB each)
  const int lane = threadIdx.x & 31;
  const int lo16 = lane & 15;
  const int hi   = (lane >> 4) & 1;
  const int wave = threadIdx.x >> 5;
  const int bh = blockIdx.y;  // b*H + h
  const int b = bh >> 4, h = bh & 15;
  const int qbase = blockIdx.x * 64 + wave * 16;

  const _Float16* qp = Q + ((size_t)bh << 11) * D_HEAD;
  const _Float16* kp = K + ((size_t)bh << 11) * D_HEAD;
  const _Float16* vp = V + ((size_t)bh << 11) * D_HEAD;
  _Float16* pw = pbuf + wave * 512;

  // Q A-fragments (two 16x32 k-slices of Dh=64): each is two 16B vector loads.
  v16h qa[2];
  {
    const _Float16* qrow = qp + (size_t)(qbase + lo16) * D_HEAD + hi * 16;
#pragma unroll
    for (int f = 0; f < 2; ++f)
      qa[f] = join8(*(const v8h*)(qrow + f * 32), *(const v8h*)(qrow + f * 32 + 8));
  }

  const v8f vzero = {0.f, 0.f, 0.f, 0.f, 0.f, 0.f, 0.f, 0.f};
  v8f o[4];
#pragma unroll
  for (int j = 0; j < 4; ++j) o[j] = vzero;
  float mrow[8], lrow[8];
#pragma unroll
  for (int r = 0; r < 8; ++r) { mrow[r] = -3.0e38f; lrow[r] = 0.f; }

  const float sc = 0.125f;  // 1/sqrt(Dh)
  const int nch = (qbase + 15) / 32 + 1;

  for (int kc = 0; kc < nch; ++kc) {
    const int k0 = kc * 32;

    // ---- S = Q @ K^T for 32 keys (two 16x16 tiles, each 2 WMMA k-steps) ----
    v8f s[2];
#pragma unroll
    for (int tt = 0; tt < 2; ++tt) {
      const _Float16* krow = kp + (size_t)(k0 + tt * 16 + lo16) * D_HEAD + hi * 16;
      v16h kb0 = *(const v16h*)(krow);       // d in [0,32)
      v16h kb1 = *(const v16h*)(krow + 32);  // d in [32,64)
      v8f acc = __builtin_amdgcn_wmma_f32_16x16x32_f16(
          false, qa[0], false, kb0, (short)0, vzero, false, false);
      acc = __builtin_amdgcn_wmma_f32_16x16x32_f16(
          false, qa[1], false, kb1, (short)0, acc, false, false);
      s[tt] = acc;
    }

    // ---- scale, causal mask, online softmax update ----
#pragma unroll
    for (int r = 0; r < 8; ++r) {
      int qrow = qbase + r + (hi << 3);
#pragma unroll
      for (int tt = 0; tt < 2; ++tt) {
        int key = k0 + tt * 16 + lo16;
        float v = s[tt][r] * sc;
        if (key > qrow) v = -3.0e38f;
        s[tt][r] = v;
      }
      float rm = xor_red_max16(fmaxf(s[0][r], s[1][r]));
      float mnew = fmaxf(mrow[r], rm);
      float corr = __expf(mrow[r] - mnew);
      lrow[r] *= corr;
#pragma unroll
      for (int j = 0; j < 4; ++j) o[j][r] *= corr;
      float p0 = __expf(s[0][r] - mnew);
      float p1 = __expf(s[1][r] - mnew);
      lrow[r] += xor_red_sum16(p0 + p1);
      mrow[r] = mnew;
      s[0][r] = p0;
      s[1][r] = p1;
    }

    // ---- transpose P (C-layout -> A-layout) through per-wave LDS ----
#pragma unroll
    for (int tt = 0; tt < 2; ++tt)
#pragma unroll
      for (int r = 0; r < 8; ++r) {
        int row = r + (hi << 3);
        pw[row * 32 + tt * 16 + lo16] = (_Float16)s[tt][r];
      }
    asm volatile("s_wait_dscnt 0" ::: "memory");  // same-wave DS store->load order
    // A-fragment read: two contiguous 8-half runs -> two ds_load_b128.
    v16h pa = join8(*(const v8h*)(pw + lo16 * 32 + hi * 16),
                    *(const v8h*)(pw + lo16 * 32 + 8 + hi * 16));

    // ---- O += P @ V (K=32 keys, 4 output 16-wide d-tiles) ----
#pragma unroll
    for (int j = 0; j < 4; ++j) {
      // True transpose gather, but single base + e*64 element stride so the 16
      // loads share a vaddr with immediate offsets.
      const _Float16* vbase = vp + (size_t)(k0 + hi * 16) * D_HEAD + j * 16 + lo16;
      v16h vb;
#pragma unroll
      for (int e = 0; e < 16; ++e) vb[e] = vbase[e * D_HEAD];
      o[j] = __builtin_amdgcn_wmma_f32_16x16x32_f16(
          false, pa, false, vb, (short)0, o[j], false, false);
    }
  }

  // ---- normalize and store f16 attn output, (B*T, D_MODEL) row-major ----
#pragma unroll
  for (int r = 0; r < 8; ++r) {
    float inv = 1.0f / lrow[r];
    int qrow = qbase + r + (hi << 3);
    size_t row = ((size_t)b << 11) + qrow;
#pragma unroll
    for (int j = 0; j < 4; ++j) {
      int col = h * D_HEAD + j * 16 + lo16;
      Oc[(row << 10) + col] = (_Float16)(o[j][r] * inv);
    }
  }
}

// ---------------------------------------------------------------------------
// Host orchestration (all launches on `stream`; no hip allocs/copies).
// Workspace layout (bytes):
//   [0,16M)    x_swz (A-packed f16)     -- reused later for attn_swz
//   [16M,24M)  wq/wk/wv/wo B-packed f16 (2MB each)
//   [24M,72M)  Q/K/V f16 in (B,H,T,Dh)  (16MB each)
//   [72M,88M)  attn f16 (B*T, D) row-major
// ---------------------------------------------------------------------------
extern "C" void kernel_launch(void* const* d_in, const int* in_sizes, int n_in,
                              void* d_out, int out_size, void* d_ws, size_t ws_size,
                              hipStream_t stream) {
  (void)in_sizes; (void)n_in; (void)out_size; (void)ws_size;
  const float* x  = (const float*)d_in[0];
  const float* wq = (const float*)d_in[1];
  const float* wk = (const float*)d_in[2];
  const float* wv = (const float*)d_in[3];
  const float* wo = (const float*)d_in[4];

  char* ws = (char*)d_ws;
  const size_t MB = 1024ull * 1024ull;
  _Float16* x_swz  = (_Float16*)(ws + 0);        // 16MB (reused as attn_swz)
  _Float16* wq_s   = (_Float16*)(ws + 16 * MB);
  _Float16* wk_s   = (_Float16*)(ws + 18 * MB);
  _Float16* wv_s   = (_Float16*)(ws + 20 * MB);
  _Float16* wo_s   = (_Float16*)(ws + 22 * MB);
  _Float16* q16    = (_Float16*)(ws + 24 * MB);
  _Float16* k16    = (_Float16*)(ws + 40 * MB);
  _Float16* v16    = (_Float16*)(ws + 56 * MB);
  _Float16* attn16 = (_Float16*)(ws + 72 * MB);

  const int PACK_X_BLOCKS = (M_ROWS * D_MODEL) / 256;     // 32768
  const int PACK_W_BLOCKS = (D_MODEL * D_MODEL) / 256;    // 4096

  pack_a_f32<<<PACK_X_BLOCKS, 256, 0, stream>>>(x, x_swz);
  pack_b_f32<<<PACK_W_BLOCKS, 256, 0, stream>>>(wq, wq_s);
  pack_b_f32<<<PACK_W_BLOCKS, 256, 0, stream>>>(wk, wk_s);
  pack_b_f32<<<PACK_W_BLOCKS, 256, 0, stream>>>(wv, wv_s);
  pack_b_f32<<<PACK_W_BLOCKS, 256, 0, stream>>>(wo, wo_s);

  dim3 gg(D_MODEL / 128, M_ROWS / 128);  // (8, 64)
  gemm_wmma<<<gg, 256, 0, stream>>>(x_swz, wq_s, nullptr, q16, 1);
  gemm_wmma<<<gg, 256, 0, stream>>>(x_swz, wk_s, nullptr, k16, 1);
  gemm_wmma<<<gg, 256, 0, stream>>>(x_swz, wv_s, nullptr, v16, 1);

  attn_flash<<<dim3(SEQ_T / 64, BATCH * N_HEADS), 128, 0, stream>>>(q16, k16, v16, attn16);

  pack_a_f16<<<PACK_X_BLOCKS, 256, 0, stream>>>(attn16, x_swz);
  gemm_wmma<<<gg, 256, 0, stream>>>(x_swz, wo_s, (float*)d_out, nullptr, 0);
}
